// QueryAdaptivePNA_89446988906927
// MI455X (gfx1250) — compile-verified, matrix operand order checked by reading, not run on hardware
//
#include <hip/hip_runtime.h>
#include <hip/hip_bf16.h>
#include <math.h>

// ---------------- problem sizes ----------------
#define Bn 1024
#define Hn 200
#define En 256
#define Rn 256
#define HT 13            // ceil(200/16)
#define HPAD 208

typedef __attribute__((ext_vector_type(16))) __bf16 v16bf;
typedef __attribute__((ext_vector_type(8)))  __bf16 v8bf;
typedef __attribute__((ext_vector_type(8)))  float  v8f;

union BF16x16 { v16bf v; unsigned short u[16]; };

__device__ __forceinline__ unsigned short f2bf_u(float f) {
    unsigned u = __builtin_bit_cast(unsigned, f);
    unsigned r = u + 0x7FFFu + ((u >> 16) & 1u);   // round-to-nearest-even
    return (unsigned short)(r >> 16);
}
__device__ __forceinline__ float bfu2f(unsigned short s) {
    unsigned u = ((unsigned)s) << 16;
    return __builtin_bit_cast(float, u);
}

#define WMMA_BF16(Af, Bf, Cf) \
    __builtin_amdgcn_wmma_f32_16x16x32_bf16(false, (Af), false, (Bf), (short)0, (Cf), false, false)

// ---- gfx1250 async global->LDS path (guarded; falls back to sync loads) ----
#if defined(__has_builtin)
#if __has_builtin(__builtin_amdgcn_global_load_async_to_lds_b128)
#define HAVE_ASYNC_LDS 1
#endif
#endif

#ifdef HAVE_ASYNC_LDS
// Builtin prototype (from compiler diagnostic): arg0 is
//   'int __attribute__((vector_size(16))) __device__ *'  (AS1 global int4*)
// so pass GCC-vector-int4 pointers in explicit address spaces.
typedef int v4i_gcc __attribute__((vector_size(16)));
typedef __attribute__((address_space(1))) v4i_gcc* as1_v4i;
typedef __attribute__((address_space(3))) v4i_gcc* as3_v4i;
__device__ __forceinline__ void async_cp16(const float* g, float* l) {
    // two-step: reinterpret in generic AS (drops const), then addrspacecast
    v4i_gcc* gg = (v4i_gcc*)g;
    v4i_gcc* ll = (v4i_gcc*)l;
    __builtin_amdgcn_global_load_async_to_lds_b128((as1_v4i)gg, (as3_v4i)ll, 0, 0);
}
#if __has_builtin(__builtin_amdgcn_s_wait_asynccnt)
#define WAIT_ASYNC(n) __builtin_amdgcn_s_wait_asynccnt(n)
#else
#define WAIT_ASYNC(n) asm volatile("s_wait_asynccnt %0" ::"i"(n))
#endif
#endif

// ---------------------------------------------------------------------------
// Kernel 0: per-row n_valid, log_d, and global mean_d (one block of 1024).
// ---------------------------------------------------------------------------
__global__ __launch_bounds__(1024)
void k_stats(const unsigned char* __restrict__ mask,
             float* __restrict__ logd, float* __restrict__ nvalid,
             float* __restrict__ meand) {
    __shared__ float red[1024];
    int b = threadIdx.x;
    const unsigned char* mrow = mask + (size_t)b * Hn;
    int cnt = 0;
    for (int h = 0; h < Hn; ++h) cnt += (mrow[h] != 0);
    float nv = fmaxf((float)cnt, 1.0f);
    nvalid[b] = nv;
    float ld = logf(nv);               // deg>=1 so log(max(deg,1)) == log(nv)
    logd[b] = ld;
    red[b] = ld;
    __syncthreads();
    for (int s = 512; s > 0; s >>= 1) {
        if (b < s) red[b] += red[b + s];
        __syncthreads();
    }
    if (b == 0) meand[0] = fmaxf(red[0] / (float)Bn, 1.0f);
}

// ---------------------------------------------------------------------------
// Generic C[M,N] = X[M,K] * W[N,K]^T (+bias[N]) via bf16 WMMA, f32 accum.
// One 16x16 output tile per wave; K must be a multiple of 32.
// ---------------------------------------------------------------------------
__global__ __launch_bounds__(128)
void k_gemm_xwt(const float* __restrict__ X, const float* __restrict__ W,
                const float* __restrict__ bias, float* __restrict__ C,
                int M, int N, int K) {
    int wave = threadIdx.x >> 5;
    int lane = threadIdx.x & 31;
    int tile = blockIdx.x * (blockDim.x >> 5) + wave;
    int ntn = N >> 4;
    int total = (M >> 4) * ntn;
    if (tile >= total) return;                 // uniform per-wave condition
    int mt = tile / ntn, nt = tile % ntn;
    int half = lane >> 4, row = lane & 15;
    const float* pa_base = X + (size_t)((mt << 4) + row) * K;
    const float* pb_base = W + (size_t)((nt << 4) + row) * K;

    v8f acc = {};
    for (int kc = 0; kc < K; kc += 32) {
        int k0 = kc + half * 8;
        BF16x16 a, bf;
        const float* pa = pa_base + k0;
        const float* pb = pb_base + k0;
#pragma unroll
        for (int j = 0; j < 8; ++j) {
            a.u[j]      = f2bf_u(pa[j]);
            a.u[8 + j]  = f2bf_u(pa[16 + j]);
            bf.u[j]     = f2bf_u(pb[j]);
            bf.u[8 + j] = f2bf_u(pb[16 + j]);
        }
        acc = WMMA_BF16(a.v, bf.v, acc);
    }
    int cn = (nt << 4) + row;
    float bv = bias ? bias[cn] : 0.0f;
#pragma unroll
    for (int v = 0; v < 8; ++v) {
        int cm = (mt << 4) + v + half * 8;
        C[(size_t)cm * N + cn] = acc[v] + bv;
    }
}

// ---------------------------------------------------------------------------
// Main fused kernel: one block per batch row b. 512 threads = 16 waves.
// Wave w owns e-tile [16w, 16w+16). Dynamic LDS layout (bytes):
//   msg   : 208*256*u16 = 106496   (off 0)
//   aT    : 16*256*u16  =   8192   (off 106496)
//   stage : 2*16*256*f32=  32768   (off 114688)  async double buffer
//   decay : 208*f32     =    832   (off 147456)
//   maskf : 208*f32     =    832   (off 148288)
//   attn  : 208*f32     =    832   (off 149120)
//   aggs  : 4*256*f32   =   4096   (off 149952)  total 154048
// ---------------------------------------------------------------------------
#define SMEM_MAIN 154048

__device__ __forceinline__ void issue_tile_async(const float* __restrict__ rel,
                                                 int b, int hb, float* stage_buf,
                                                 int tid) {
#ifdef HAVE_ASYNC_LDS
    // 16 rows x 256 f32 = 1024 chunks of 16B; 512 threads -> 2 async ops each.
    // Uniform issue: pad rows clamp the *source* row (dst contents unused).
#pragma unroll
    for (int s = 0; s < 2; ++s) {
        int c = tid + s * 512;
        int r = c >> 6;
        int h = hb + r;
        if (h > Hn - 1) h = Hn - 1;            // clamp: keep EXEC full / uniform
        int cc = (c & 63) * 4;
        const float* g = rel + ((size_t)b * Hn + h) * Rn + cc;
        async_cp16(g, stage_buf + c * 4);
    }
#endif
}

__global__ __launch_bounds__(512)
void k_main(const float* __restrict__ e_emb, const float* __restrict__ nb_emb,
            const float* __restrict__ rel, const float* __restrict__ delta_t,
            const unsigned char* __restrict__ mask, const float* __restrict__ W_rel,
            const float* __restrict__ log_gamma, const float* __restrict__ qkey,
            const float* __restrict__ logd_g, const float* __restrict__ nvalid_g,
            const float* __restrict__ meand_g, float* __restrict__ feats) {
    extern __shared__ char smem[];
    unsigned short* msg  = (unsigned short*)smem;
    unsigned short* aT   = (unsigned short*)(smem + 106496);
    float* stage = (float*)(smem + 114688);
    float* decay = (float*)(smem + 147456);
    float* maskf = (float*)(smem + 148288);
    float* attn  = (float*)(smem + 149120);
    float* aggS  = (float*)(smem + 149952);
    float* aggMx = aggS + 256;
    float* aggMn = aggMx + 256;
    float* aggSq = aggMn + 256;

    const int b    = blockIdx.x;
    const int tid  = threadIdx.x;
    const int wave = tid >> 5, lane = tid & 31;
    const int half = lane >> 4, col = lane & 15;
    const int e_base = wave << 4;

    const float gamma = expf(log_gamma[0]);

    // per-h decay / mask / attn init
    if (tid < HPAD) {
        int h = tid;
        if (h < Hn) {
            float dt = delta_t[(size_t)b * Hn + h];
            decay[h] = expf(-gamma * fmaxf(dt, 0.0f));
            maskf[h] = (mask[(size_t)b * Hn + h] != 0) ? 1.0f : 0.0f;
        } else {
            decay[h] = 0.0f;
            maskf[h] = 0.0f;
        }
        attn[h] = 0.0f;
    }

#ifdef HAVE_ASYNC_LDS
    issue_tile_async(rel, b, 0, stage, tid);   // preload tile 0
#endif

    // B fragments (W_rel rows for this wave's e-tile), kept in VGPRs for all H
    v16bf bw[8];
    {
        const float* wr = W_rel + (size_t)(e_base + col) * Rn;
#pragma unroll
        for (int kc = 0; kc < 8; ++kc) {
            BF16x16 t;
            const float* p = wr + kc * 32 + half * 8;
#pragma unroll
            for (int j = 0; j < 8; ++j) {
                t.u[j]     = f2bf_u(p[j]);
                t.u[8 + j] = f2bf_u(p[16 + j]);
            }
            bw[kc] = t.v;
        }
    }
    const float qk = qkey[(size_t)b * En + e_base + col];

    float sumv = 0.0f, sqv = 0.0f, mxv = -1e30f, mnv = 1e4f;
    __syncthreads();

    for (int ht = 0; ht < HT; ++ht) {
        const int hb = ht << 4;
        const int buf = ht & 1;

#ifdef HAVE_ASYNC_LDS
        // prefetch next tile into the other staging buffer, then drain current
        if (ht + 1 < HT) {
            issue_tile_async(rel, b, hb + 16, stage + (buf ^ 1) * 4096, tid);
            WAIT_ASYNC(2);                      // my 2 current-tile ops done
        } else {
            WAIT_ASYNC(0);
        }
        __syncthreads();                        // all waves' stage[buf] visible
        // convert stage[buf] f32 -> bf16 aT (pad rows forced to zero)
        for (int i = tid; i < 16 * 256; i += 512) {
            int h = hb + (i >> 8);
            float v = (h < Hn) ? stage[buf * 4096 + i] : 0.0f;
            aT[i] = f2bf_u(v);
        }
#else
        // fallback: synchronous stage of the bf16 A tile
        for (int i = tid; i < 16 * 256; i += 512) {
            int r = i >> 8, c = i & 255;
            int h = hb + r;
            float v = (h < Hn) ? rel[((size_t)b * Hn + h) * Rn + c] : 0.0f;
            aT[i] = f2bf_u(v);
        }
#endif
        __syncthreads();

        // rel_proj 16x16 tile for this wave
        v8f acc = {};
#pragma unroll
        for (int kc = 0; kc < 8; ++kc) {
            const v8bf* plo = (const v8bf*)(aT + col * 256 + kc * 32 + half * 8);
            const v8bf* phi = (const v8bf*)(aT + col * 256 + kc * 32 + 16 + half * 8);
            v8bf lo = plo[0], hi = phi[0];
            v16bf af = __builtin_shufflevector(lo, hi,
                0, 1, 2, 3, 4, 5, 6, 7, 8, 9, 10, 11, 12, 13, 14, 15);
            acc = WMMA_BF16(af, bw[kc], acc);
        }

        // messages + running aggregates  (lane: e = e_base+col, VGPR v: h = hb+v+8*half)
        float p[8];
#pragma unroll
        for (int v = 0; v < 8; ++v) {
            int h = hb + v + (half << 3);
            bool inr = (h < Hn);
            float nb = inr ? nb_emb[((size_t)b * Hn + h) * En + e_base + col] : 0.0f;
            if (inr && h + 16 < Hn)             // hint next tile of the nb stream
                __builtin_prefetch(&nb_emb[((size_t)b * Hn + h + 16) * En + e_base + col], 0, 0);
            float md = acc[v] * nb * (inr ? decay[h] : 0.0f);
            float mf = inr ? maskf[h] : 0.0f;
            float mv = md * mf;
            if (inr) {
                sumv += mv;
                float c10 = fminf(fmaxf(mv, -10.0f), 10.0f);
                sqv += c10 * c10;
                mxv = fmaxf(mxv, mv);                       // masked -> 0 participates
                mnv = fminf(mnv, (mf > 0.0f) ? md : 1e4f);
                msg[h * 256 + e_base + col] = f2bf_u(md);   // store messages_dec (unmasked)
            }
            p[v] = inr ? md * qk : 0.0f;
        }

        // attn_raw[h] partials: reduce across the 16 lanes of each half
#pragma unroll
        for (int v = 0; v < 8; ++v) {
            float x = p[v];
            x += __shfl_xor(x, 8, 16);
            x += __shfl_xor(x, 4, 16);
            x += __shfl_xor(x, 2, 16);
            x += __shfl_xor(x, 1, 16);
            if (col == 0) {
                int h = hb + v + (half << 3);
                if (h < Hn) atomicAdd(&attn[h], x);
            }
        }
        __syncthreads();
    }

    // fold lane halves (same e, complementary h sets), publish per-e aggregates
    sumv += __shfl_down(sumv, 16);
    sqv  += __shfl_down(sqv, 16);
    mxv   = fmaxf(mxv, __shfl_down(mxv, 16));
    mnv   = fminf(mnv, __shfl_down(mnv, 16));
    if (half == 0) {
        aggS[e_base + col]  = sumv;
        aggMx[e_base + col] = mxv;
        aggMn[e_base + col] = mnv;
        aggSq[e_base + col] = sqv;
    }
    __syncthreads();

    // softmax over attn (wave 0)
    if (wave == 0) {
        float lm = -1e30f;
        for (int h = lane; h < Hn; h += 32) {
            float r = attn[h] * 0.0625f;                    // 1/sqrt(256)
            r = (maskf[h] > 0.0f) ? r : -10000.0f;
            attn[h] = r;
            lm = fmaxf(lm, r);
        }
        for (int s = 16; s > 0; s >>= 1) lm = fmaxf(lm, __shfl_xor(lm, s, 32));
        float ls = 0.0f;
        for (int h = lane; h < Hn; h += 32) {
            float e = expf(attn[h] - lm);
            attn[h] = e;
            ls += e;
        }
        for (int s = 16; s > 0; s >>= 1) ls += __shfl_xor(ls, s, 32);
        float inv = 1.0f / ls;
        for (int h = lane; h < Hn; h += 32) attn[h] *= inv;
    }
    __syncthreads();

    // finalize: attn_v second pass + feats row
    if (tid < En) {
        int e = tid;
        float n    = nvalid_g[b];
        float ld   = logd_g[b];
        float mdn  = meand_g[0];
        float mean = aggS[e] / n;
        float sqm  = aggSq[e] / n;
        float stdv = sqrtf(fmaxf(sqm - mean * mean, 1e-6f));
        float mx   = aggMx[e];
        float mn   = fminf(fmaxf(aggMn[e], -1e4f), 1e4f);
        float av = 0.0f;
        for (int h = 0; h < Hn; ++h) av += attn[h] * bfu2f(msg[h * 256 + e]);
        float sd = ld / mdn;
        float si = mdn / fmaxf(ld, 0.01f);
        float* fr = feats + (size_t)b * (14 * En);
        fr[0 * En + e]  = mean;      fr[1 * En + e]  = mean * sd; fr[2 * En + e]  = mean * si;
        fr[3 * En + e]  = mx;        fr[4 * En + e]  = mx * sd;   fr[5 * En + e]  = mx * si;
        fr[6 * En + e]  = mn;        fr[7 * En + e]  = mn * sd;   fr[8 * En + e]  = mn * si;
        fr[9 * En + e]  = stdv;      fr[10 * En + e] = stdv * sd; fr[11 * En + e] = stdv * si;
        fr[12 * En + e] = av;
        fr[13 * En + e] = e_emb[(size_t)b * En + e];
    }
}

// ---------------------------------------------------------------------------
// Tail: LN -> GELU(exact) -> W2^T -> LN ; one 256-thread block per row.
// ---------------------------------------------------------------------------
__device__ __forceinline__ float block_sum256(float x, float* red, int t) {
    for (int s = 16; s > 0; s >>= 1) x += __shfl_xor(x, s, 32);
    if ((t & 31) == 0) red[t >> 5] = x;
    __syncthreads();
    if (t == 0) {
        float a = 0.0f;
        for (int i = 0; i < 8; ++i) a += red[i];
        red[0] = a;
    }
    __syncthreads();
    float tot = red[0];
    __syncthreads();
    return tot;
}

__global__ __launch_bounds__(256)
void k_tail(const float* __restrict__ G, const float* __restrict__ g1,
            const float* __restrict__ be1, const float* __restrict__ W2,
            const float* __restrict__ b2, const float* __restrict__ g2,
            const float* __restrict__ be2, float* __restrict__ out) {
    __shared__ float hrow[512];
    __shared__ float red[8];
    int b = blockIdx.x, t = threadIdx.x;
    float x0 = G[(size_t)b * 512 + t];
    float x1 = G[(size_t)b * 512 + 256 + t];
    float mean = block_sum256(x0 + x1, red, t) * (1.0f / 512.0f);
    float d0 = x0 - mean, d1 = x1 - mean;
    float var = block_sum256(d0 * d0 + d1 * d1, red, t) * (1.0f / 512.0f);
    float inv = rsqrtf(var + 1e-5f);
    float y0 = d0 * inv * g1[t] + be1[t];
    float y1 = d1 * inv * g1[256 + t] + be1[256 + t];
    y0 = 0.5f * y0 * (1.0f + erff(y0 * 0.70710678f));
    y1 = 0.5f * y1 * (1.0f + erff(y1 * 0.70710678f));
    hrow[t] = y0;
    hrow[256 + t] = y1;
    __syncthreads();
    float acc = b2[t];
    const float* w = W2 + (size_t)t * 512;
    for (int k = 0; k < 512; ++k) acc += hrow[k] * w[k];
    float m2 = block_sum256(acc, red, t) * (1.0f / 256.0f);
    float d = acc - m2;
    float v2 = block_sum256(d * d, red, t) * (1.0f / 256.0f);
    out[(size_t)b * En + t] = d * rsqrtf(v2 + 1e-5f) * g2[t] + be2[t];
}

// ---------------------------------------------------------------------------
extern "C" void kernel_launch(void* const* d_in, const int* in_sizes, int n_in,
                              void* d_out, int out_size, void* d_ws, size_t ws_size,
                              hipStream_t stream) {
    const float* e_emb       = (const float*)d_in[0];
    const float* nb_emb      = (const float*)d_in[1];
    const float* nb_rel_emb  = (const float*)d_in[2];
    const float* delta_t     = (const float*)d_in[3];
    const float* query_emb   = (const float*)d_in[4];
    const unsigned char* hist_mask = (const unsigned char*)d_in[5];
    const float* W_rel       = (const float*)d_in[6];
    const float* W_attn      = (const float*)d_in[7];
    const float* log_gamma   = (const float*)d_in[8];
    const float* W1          = (const float*)d_in[9];
    const float* b1          = (const float*)d_in[10];
    const float* g1          = (const float*)d_in[11];
    const float* be1         = (const float*)d_in[12];
    const float* W2          = (const float*)d_in[13];
    const float* b2          = (const float*)d_in[14];
    const float* g2          = (const float*)d_in[15];
    const float* be2         = (const float*)d_in[16];
    float* out = (float*)d_out;

    float* ws     = (float*)d_ws;
    float* qkey   = ws;                           // B*E        = 262144
    float* logd   = qkey + (size_t)Bn * En;       // B
    float* nvalid = logd + Bn;                    // B
    float* meand  = nvalid + Bn;                  // 1 (+pad 16)
    float* feats  = meand + 16;                   // B*14E      = 3670016
    float* G      = feats + (size_t)Bn * 14 * En; // B*512      = 524288

    // 1) per-row stats + global mean_d
    k_stats<<<1, 1024, 0, stream>>>(hist_mask, logd, nvalid, meand);

    // 2) q_key = query_emb @ W_attn^T  (M=1024,N=256,K=256): 1024 tiles / 4 waves
    k_gemm_xwt<<<256, 128, 0, stream>>>(query_emb, W_attn, nullptr, qkey,
                                        Bn, En, En);

    // 3) fused message GEMM + PNA aggregations + attention -> feats
    k_main<<<Bn, 512, SMEM_MAIN, stream>>>(e_emb, nb_emb, nb_rel_emb, delta_t,
                                           hist_mask, W_rel, log_gamma, qkey,
                                           logd, nvalid, meand, feats);

    // 4) G = feats @ W1^T + b1  (M=1024,N=512,K=3584): 2048 tiles / 4 waves
    k_gemm_xwt<<<512, 128, 0, stream>>>(feats, W1, b1, G, Bn, 512, 14 * En);

    // 5) LN -> GELU -> W2^T -> LN
    k_tail<<<Bn, 256, 0, stream>>>(G, g1, be1, W2, b2, g2, be2, out);
}
